// SegmentClassifier_28011776705008
// MI455X (gfx1250) — matrix operand order repaired and checked by reading, not compiled, not run on hardware
//
#include <hip/hip_runtime.h>
#include <hip/hip_bf16.h>
#include <stdint.h>

#define NN 4096      // nodes
#define NE 12288     // edges
#define DC 11        // combined feature dim
#define DP 16        // padded feature dim

typedef __attribute__((ext_vector_type(16))) _Float16 v16h;
typedef __attribute__((ext_vector_type(8)))  _Float16 v8h;
typedef __attribute__((ext_vector_type(8)))  float    v8f;

// ---------------------------------------------------------------------------
// Convert fp32 [NN,NE] -> fp16 natural [NN,NE] and fp16 transposed [NE,NN]
// ---------------------------------------------------------------------------
__global__ void cvt_transpose_kernel(const float* __restrict__ src,
                                     _Float16* __restrict__ dstN,
                                     _Float16* __restrict__ dstT)
{
    __shared__ float tile[32][33];
    int tx = threadIdx.x & 31;
    int ty = threadIdx.x >> 5;          // 0..7
    int e0 = blockIdx.x * 32;
    int n0 = blockIdx.y * 32;
    #pragma unroll
    for (int r = 0; r < 32; r += 8) {
        int n = n0 + ty + r;
        float v = src[(size_t)n * NE + e0 + tx];
        tile[ty + r][tx] = v;
        dstN[(size_t)n * NE + e0 + tx] = (_Float16)v;
    }
    __syncthreads();
    #pragma unroll
    for (int r = 0; r < 32; r += 8) {
        int e = e0 + ty + r;
        dstT[(size_t)e * NN + n0 + tx] = (_Float16)tile[tx][ty + r];
    }
}

// ---------------------------------------------------------------------------
// X init: H = tanh(nodes @ W_in + b_in);  X = [H, nodes, 0-pad] as [NN,16] f32
// ---------------------------------------------------------------------------
__global__ void init_x_kernel(const float* __restrict__ nodes,
                              const float* __restrict__ W_in,
                              const float* __restrict__ b_in,
                              float* __restrict__ X)
{
    int n = blockIdx.x * blockDim.x + threadIdx.x;
    if (n >= NN) return;
    float i0 = nodes[n*3+0], i1 = nodes[n*3+1], i2 = nodes[n*3+2];
    float xr[DP];
    #pragma unroll
    for (int j = 0; j < 8; ++j) {
        float s = b_in[j] + i0*W_in[0*8+j] + i1*W_in[1*8+j] + i2*W_in[2*8+j];
        xr[j] = tanhf(s);
    }
    xr[8] = i0; xr[9] = i1; xr[10] = i2;
    xr[11] = xr[12] = xr[13] = xr[14] = xr[15] = 0.f;
    #pragma unroll
    for (int j = 0; j < DP; ++j) X[(size_t)n*DP + j] = xr[j];
}

// ---------------------------------------------------------------------------
// Pack a dense [K x 16] f32 operand into WMMA B-fragment order (f16):
//   frag dword index = kb*256 + lane*8 + v
//   lane<16 : col = lane,    K = kb*32 + 2v, 2v+1
//   lane>=16: col = lane-16, K = kb*32 + 16 + 2v, 16+2v+1
// ---------------------------------------------------------------------------
__global__ void pack_frag_kernel(const float* __restrict__ S,
                                 uint32_t* __restrict__ frag,
                                 int kblocks)
{
    int t = blockIdx.x * blockDim.x + threadIdx.x;
    if (t >= kblocks * 256) return;
    int kb   = t >> 8;
    int r    = t & 255;
    int lane = r >> 3;
    int v    = r & 7;
    int col  = lane & 15;
    int k0   = kb*32 + ((lane >> 4) << 4) + 2*v;
    union { _Float16 h[2]; uint32_t u; } cv;
    cv.h[0] = (_Float16)S[(size_t)k0      * DP + col];
    cv.h[1] = (_Float16)S[(size_t)(k0+1)  * DP + col];
    frag[t] = cv.u;
}

// ---------------------------------------------------------------------------
// C[M x 16] (k-split partials) = A[M x K] (f16 row-major) * Bfrag
// One 16-row tile per wave; 8 waves per block; K split over blockIdx.y.
// B-fragment chunk staged once into LDS (shared by all 8 waves).
// A loads software-pipelined; two accumulators break the WMMA dep chain.
// ---------------------------------------------------------------------------
template <int CHUNK>   // number of 32-deep K blocks handled by this block
__global__ __launch_bounds__(256)
void gemm_wmma_kernel(const _Float16* __restrict__ A,
                      const uint32_t* __restrict__ Bfrag,
                      float* __restrict__ Cpart,
                      int M, int K)
{
    __shared__ uint32_t ldsB[CHUNK * 256];

    int tid  = threadIdx.x;
    int wave = tid >> 5;
    int lane = tid & 31;
    int kb0  = blockIdx.y * CHUNK;

    // cooperative stage of the B-fragment chunk (coalesced b32 per thread)
    const uint32_t* bsrc = Bfrag + (size_t)kb0 * 256;
    #pragma unroll
    for (int i = 0; i < CHUNK; ++i)
        ldsB[i*256 + tid] = bsrc[i*256 + tid];
    __syncthreads();

    int tile = blockIdx.x * 8 + wave;
    int m0   = tile * 16;
    if (m0 >= M) return;                       // wave-uniform

    const _Float16* arow = A + (size_t)(m0 + (lane & 15)) * K + (size_t)kb0 * 32
                             + ((lane < 16) ? 0 : 16);

    v8f acc0 = {0.f,0.f,0.f,0.f,0.f,0.f,0.f,0.f};
    v8f acc1 = {0.f,0.f,0.f,0.f,0.f,0.f,0.f,0.f};

    // prologue loads
    v8h a0 = *(const v8h*)(arow);
    v8h a1 = *(const v8h*)(arow + 8);

    #pragma unroll
    for (int j = 0; j < CHUNK; ++j) {
        v8h n0, n1;
        if (j + 1 < CHUNK) {                   // prefetch next A fragment
            const _Float16* np = arow + (size_t)(j + 1) * 32;
            n0 = *(const v8h*)(np);
            n1 = *(const v8h*)(np + 8);
        }
        v16h a;
        #pragma unroll
        for (int i = 0; i < 8; ++i) { a[i] = a0[i]; a[i+8] = a1[i]; }
        v16h b = *(const v16h*)(&ldsB[j*256 + lane*8]);
        if (j & 1)
            acc1 = __builtin_amdgcn_wmma_f32_16x16x32_f16(
                       false, a, false, b, (short)0, acc1, false, false);
        else
            acc0 = __builtin_amdgcn_wmma_f32_16x16x32_f16(
                       false, a, false, b, (short)0, acc0, false, false);
        a0 = n0; a1 = n1;
    }

    float* cp = Cpart + (size_t)blockIdx.y * M * DP;
    int rbase = m0 + ((lane >> 4) << 3);
    int ccol  = lane & 15;
    #pragma unroll
    for (int v = 0; v < 8; ++v)
        cp[(size_t)(rbase + v) * DP + ccol] = acc0[v] + acc1[v];
}

// ---------------------------------------------------------------------------
// Edge MLP: reduce bo/bi partials, e = sigmoid(tanh([bo,bi]@We1+be1)@We2+be2)
// Non-final: write ebo = e*bo, ebi = e*bi (padded [NE,16] f32).
// Final: write e to output.
// ---------------------------------------------------------------------------
__global__ void edge_mlp_kernel(const float* __restrict__ boP,
                                const float* __restrict__ biP, int nParts,
                                const float* __restrict__ We1, const float* __restrict__ be1,
                                const float* __restrict__ We2, const float* __restrict__ be2,
                                float* __restrict__ ebo, float* __restrict__ ebi,
                                float* __restrict__ eOut, int writeOut)
{
    int e = blockIdx.x * blockDim.x + threadIdx.x;
    if (e >= NE) return;
    float bo[DC], bi[DC];
    #pragma unroll
    for (int d = 0; d < DC; ++d) {
        float s = 0.f, t = 0.f;
        for (int p = 0; p < nParts; ++p) {
            s += boP[(size_t)p*NE*DP + (size_t)e*DP + d];
            t += biP[(size_t)p*NE*DP + (size_t)e*DP + d];
        }
        bo[d] = s; bi[d] = t;
    }
    float h[8];
    #pragma unroll
    for (int j = 0; j < 8; ++j) {
        float s = be1[j];
        #pragma unroll
        for (int i = 0; i < DC; ++i) s += bo[i] * We1[i*8 + j];
        #pragma unroll
        for (int i = 0; i < DC; ++i) s += bi[i] * We1[(DC+i)*8 + j];
        h[j] = tanhf(s);
    }
    float ev = be2[0];
    #pragma unroll
    for (int j = 0; j < 8; ++j) ev += h[j] * We2[j];
    ev = 1.f / (1.f + expf(-ev));
    if (writeOut) { eOut[e] = ev; return; }
    #pragma unroll
    for (int d = 0; d < DP; ++d) {
        ebo[(size_t)e*DP + d] = (d < DC) ? ev * bo[d] : 0.f;
        ebi[(size_t)e*DP + d] = (d < DC) ? ev * bi[d] : 0.f;
    }
}

// ---------------------------------------------------------------------------
// Node MLP + residual: M=[mi,mo,X]; X += [tanh(tanh(M@Wn1+bn1)@Wn2+bn2), nodes]
// ---------------------------------------------------------------------------
__global__ void node_update_kernel(const float* __restrict__ miP,
                                   const float* __restrict__ moP, int nParts,
                                   const float* __restrict__ nodes,
                                   const float* __restrict__ Wn1, const float* __restrict__ bn1,
                                   const float* __restrict__ Wn2, const float* __restrict__ bn2,
                                   float* __restrict__ X)
{
    int n = blockIdx.x * blockDim.x + threadIdx.x;
    if (n >= NN) return;
    float M[33];
    #pragma unroll
    for (int d = 0; d < DC; ++d) {
        float s = 0.f, t = 0.f;
        for (int p = 0; p < nParts; ++p) {
            s += miP[(size_t)p*NN*DP + (size_t)n*DP + d];
            t += moP[(size_t)p*NN*DP + (size_t)n*DP + d];
        }
        M[d] = s; M[DC + d] = t; M[2*DC + d] = X[(size_t)n*DP + d];
    }
    float h1[8];
    #pragma unroll
    for (int j = 0; j < 8; ++j) {
        float s = bn1[j];
        #pragma unroll
        for (int i = 0; i < 33; ++i) s += M[i] * Wn1[i*8 + j];
        h1[j] = tanhf(s);
    }
    float h2[8];
    #pragma unroll
    for (int j = 0; j < 8; ++j) {
        float s = bn2[j];
        #pragma unroll
        for (int i = 0; i < 8; ++i) s += h1[i] * Wn2[i*8 + j];
        h2[j] = tanhf(s);
    }
    #pragma unroll
    for (int j = 0; j < 8; ++j) X[(size_t)n*DP + j] = M[2*DC + j] + h2[j];
    #pragma unroll
    for (int k = 0; k < 3; ++k) X[(size_t)n*DP + 8 + k] = M[2*DC + 8 + k] + nodes[n*3 + k];
}

// ---------------------------------------------------------------------------
extern "C" void kernel_launch(void* const* d_in, const int* in_sizes, int n_in,
                              void* d_out, int out_size, void* d_ws, size_t ws_size,
                              hipStream_t stream)
{
    const float* nodes = (const float*)d_in[0];
    const float* Ri    = (const float*)d_in[1];
    const float* Ro    = (const float*)d_in[2];
    const float* W_in  = (const float*)d_in[3];
    const float* b_in  = (const float*)d_in[4];
    const float* We1   = (const float*)d_in[5];
    const float* be1   = (const float*)d_in[6];
    const float* We2   = (const float*)d_in[7];
    const float* be2   = (const float*)d_in[8];
    const float* Wn1   = (const float*)d_in[9];
    const float* bn1   = (const float*)d_in[10];
    const float* Wn2   = (const float*)d_in[11];
    const float* bn2   = (const float*)d_in[12];
    float* out = (float*)d_out;

    char* ws = (char*)d_ws;
    size_t off = 0;
    auto alloc = [&](size_t bytes) -> char* {
        char* p = ws + off;
        off += (bytes + 255) & ~(size_t)255;
        return p;
    };

    const int KSPLIT_E = 4;   // k-split for bo/bi (K = NN)  -> CHUNK 32
    const int KSPLIT_N = 8;   // k-split for mi/mo (K = NE)  -> CHUNK 48

    _Float16* RoT = (_Float16*)alloc((size_t)NE * NN * 2);   // Ro^T f16 [NE,NN]
    _Float16* RiT = (_Float16*)alloc((size_t)NE * NN * 2);   // Ri^T f16 [NE,NN]
    _Float16* RoN = (_Float16*)alloc((size_t)NN * NE * 2);   // Ro   f16 [NN,NE]
    _Float16* RiN = (_Float16*)alloc((size_t)NN * NE * 2);   // Ri   f16 [NN,NE]
    float*    X   = (float*)   alloc((size_t)NN * DP * 4);   // node state, padded
    uint32_t* XB  = (uint32_t*)alloc((size_t)(NN/32) * 256 * 4);
    float*    boP = (float*)   alloc((size_t)KSPLIT_E * NE * DP * 4);
    float*    biP = (float*)   alloc((size_t)KSPLIT_E * NE * DP * 4);
    float*    ebo = (float*)   alloc((size_t)NE * DP * 4);
    float*    ebi = (float*)   alloc((size_t)NE * DP * 4);
    uint32_t* EBo = (uint32_t*)alloc((size_t)(NE/32) * 256 * 4);
    uint32_t* EBi = (uint32_t*)alloc((size_t)(NE/32) * 256 * 4);
    float*    miP = (float*)   alloc((size_t)KSPLIT_N * NN * DP * 4);
    float*    moP = (float*)   alloc((size_t)KSPLIT_N * NN * DP * 4);

    // 1) one-time fp32 -> fp16 conversion (natural + transposed layouts)
    dim3 tgrid(NE/32, NN/32);
    cvt_transpose_kernel<<<tgrid, 256, 0, stream>>>(Ri, RiN, RiT);
    cvt_transpose_kernel<<<tgrid, 256, 0, stream>>>(Ro, RoN, RoT);

    // 2) X init
    init_x_kernel<<<NN/256, 256, 0, stream>>>(nodes, W_in, b_in, X);

    // 3) message-passing iterations (3 full + final edge net)
    for (int it = 0; it < 4; ++it) {
        pack_frag_kernel<<<(NN/32), 256, 0, stream>>>(X, XB, NN/32);

        // bo = Ro^T X ; bi = Ri^T X   (M = NE, K = NN), CHUNK = 128/4 = 32
        gemm_wmma_kernel<32><<<dim3(NE/16/8, KSPLIT_E), 256, 0, stream>>>(
            RoT, XB, boP, NE, NN);
        gemm_wmma_kernel<32><<<dim3(NE/16/8, KSPLIT_E), 256, 0, stream>>>(
            RiT, XB, biP, NE, NN);

        int final_pass = (it == 3);
        edge_mlp_kernel<<<NE/256, 256, 0, stream>>>(
            boP, biP, KSPLIT_E, We1, be1, We2, be2, ebo, ebi, out, final_pass);
        if (final_pass) break;

        pack_frag_kernel<<<(NE/32), 256, 0, stream>>>(ebo, EBo, NE/32);
        pack_frag_kernel<<<(NE/32), 256, 0, stream>>>(ebi, EBi, NE/32);

        // mi = Ri (e.bo) ; mo = Ro (e.bi)   (M = NN, K = NE), CHUNK = 384/8 = 48
        gemm_wmma_kernel<48><<<dim3(NN/16/8, KSPLIT_N), 256, 0, stream>>>(
            RiN, EBo, miP, NN, NE);
        gemm_wmma_kernel<48><<<dim3(NN/16/8, KSPLIT_N), 256, 0, stream>>>(
            RoN, EBi, moP, NN, NE);

        node_update_kernel<<<NN/256, 256, 0, stream>>>(
            miP, moP, KSPLIT_N, nodes, Wn1, bn1, Wn2, bn2, X);
    }
    (void)in_sizes; (void)n_in; (void)out_size; (void)ws_size;
}
